// NodeUpdateNetwork_46677704573340
// MI455X (gfx1250) — matrix operand order repaired and checked by reading, not compiled, not run on hardware
//
#include <hip/hip_runtime.h>

// ---------------------------------------------------------------------------
// NodeUpdateNetwork on MI455X (gfx1250, wave32, WMMA)
// B=16, N=2048, F=64.  Memory-bound by the 512MB edge_feats read (~23us floor
// at 23.3 TB/s).  Both GEMMs run on v_wmma_f32_16x16x32_bf16 with f32 accum.
// edge_feats loads are non-temporal (streamed once); Xt / W / z stay temporal
// so the 192MB L2 keeps the reused operands hot.
// ---------------------------------------------------------------------------

#define Bz 16
#define Nz 2048
#define Fz 64
#define BN_EPS 1e-5f

typedef __attribute__((ext_vector_type(16))) __bf16 v16bf;
typedef __attribute__((ext_vector_type(8)))  __bf16 v8bf;
typedef __attribute__((ext_vector_type(8)))  float  v8f;
typedef __attribute__((ext_vector_type(4)))  float  f4;

static __device__ __forceinline__ v16bf cat8(v8bf lo, v8bf hi) {
  return __builtin_shufflevector(lo, hi, 0, 1, 2, 3, 4, 5, 6, 7,
                                         8, 9, 10, 11, 12, 13, 14, 15);
}

// ---------------------------------------------------------------------------
// k0a: X (B,N,F) f32 row-major  ->  Xt (B,F,N) bf16   (makes WMMA B-tiles of
// the first GEMM contiguous along K).  1024 blocks x 256 thr, 32-row tiles.
// ---------------------------------------------------------------------------
__global__ __launch_bounds__(256) void k_transpose_x(
    const float* __restrict__ x, __bf16* __restrict__ xt) {
  __shared__ float tile[32][65];  // +1 pad: conflict-free column reads
  const int b  = blockIdx.x >> 6;         // 64 j-tiles per batch
  const int j0 = (blockIdx.x & 63) << 5;  // *32
  const int tid = threadIdx.x;
  const float* src = x + ((size_t)b * Nz + j0) * Fz;
#pragma unroll
  for (int e = 0; e < 8; ++e) {
    int lin = tid + 256 * e;      // 0..2047
    int j = lin >> 6, f = lin & 63;
    tile[j][f] = src[(size_t)j * Fz + f];     // coalesced row reads
  }
  __syncthreads();
  __bf16* dst = xt + (size_t)b * Fz * Nz;
#pragma unroll
  for (int e = 0; e < 8; ++e) {
    int lin = tid + 256 * e;
    int f = lin >> 5, j = lin & 31;
    dst[(size_t)f * Nz + j0 + j] = (__bf16)tile[j][f];  // coalesced col writes
  }
}

// k0b: W (F,F) f32 -> bf16 (row-major kept: rows are K-contiguous for GEMM2 B)
__global__ __launch_bounds__(256) void k_convert_w(
    const float* __restrict__ w, __bf16* __restrict__ wb) {
  int idx = blockIdx.x * 256 + threadIdx.x;
  if (idx < Fz * Fz) wb[idx] = (__bf16)w[idx];
}

// ---------------------------------------------------------------------------
// k1: per wave: 16 rows of one batch.
//   acc  = (S - D) @ X            (K = 2048, steps of 32, 4 N-tiles)
//   y    = x + acc/N              (C layout)  -> bf16 via LDS -> A layout
//   z    = y @ W^T                (K = 64, 8 WMMAs)
//   store z, emit per-WG per-channel (sum, sumsq) partials (deterministic).
// ---------------------------------------------------------------------------
__global__ __launch_bounds__(256) void k_fused_gemm(
    const float* __restrict__ node, const float* __restrict__ edge,
    const __bf16* __restrict__ xt, const __bf16* __restrict__ wb,
    float* __restrict__ z, float* __restrict__ psum, float* __restrict__ psq) {
  __shared__ __bf16 ldsY[8 * 16 * 64];  // 16KB: per-wave y tiles (16x64 bf16)
  __shared__ float  ldsSum[8 * 64];
  __shared__ float  ldsSq[8 * 64];

  const int b  = blockIdx.x >> 4;
  const int rb = blockIdx.x & 15;
  const int wv = threadIdx.x >> 5;
  const int ln = threadIdx.x & 31;
  const int g  = ln >> 4;      // half-wave group (WMMA layout)
  const int n  = ln & 15;      // col-in-tile (B/C layout) == row (A layout)
  const int m0 = rb * 128 + wv * 16;

  // A operand: this lane's row of S and D (A layout: lane = row).
  const float* Sp = edge + ((size_t)b * 2) * (size_t)Nz * Nz + (size_t)(m0 + n) * Nz;
  const float* Dp = Sp + (size_t)Nz * Nz;
  const __bf16* xtb = xt + (size_t)b * Fz * Nz;

  v8f acc[4] = {v8f{}, v8f{}, v8f{}, v8f{}};

#pragma unroll 2
  for (int k0 = 0; k0 < Nz; k0 += 32) {
    // A 16x32 bf16 layout: g=0 -> K {0..7, 16..23}; g=1 -> K {8..15, 24..31}
    const int c1 = k0 + 8 * g;       // chunk1 base
    const int c2 = c1 + 16;          // chunk2 base
    // edge_feats is streamed exactly once: non-temporal so it doesn't evict
    // the reused Xt / z working set from L2.
    const f4 s0 = __builtin_nontemporal_load((const f4*)(Sp + c1));
    const f4 s1 = __builtin_nontemporal_load((const f4*)(Sp + c1 + 4));
    const f4 s2 = __builtin_nontemporal_load((const f4*)(Sp + c2));
    const f4 s3 = __builtin_nontemporal_load((const f4*)(Sp + c2 + 4));
    const f4 d0 = __builtin_nontemporal_load((const f4*)(Dp + c1));
    const f4 d1 = __builtin_nontemporal_load((const f4*)(Dp + c1 + 4));
    const f4 d2 = __builtin_nontemporal_load((const f4*)(Dp + c2));
    const f4 d3 = __builtin_nontemporal_load((const f4*)(Dp + c2 + 4));
    v16bf a;
    a[0]  = (__bf16)(s0[0] - d0[0]); a[1]  = (__bf16)(s0[1] - d0[1]);
    a[2]  = (__bf16)(s0[2] - d0[2]); a[3]  = (__bf16)(s0[3] - d0[3]);
    a[4]  = (__bf16)(s1[0] - d1[0]); a[5]  = (__bf16)(s1[1] - d1[1]);
    a[6]  = (__bf16)(s1[2] - d1[2]); a[7]  = (__bf16)(s1[3] - d1[3]);
    a[8]  = (__bf16)(s2[0] - d2[0]); a[9]  = (__bf16)(s2[1] - d2[1]);
    a[10] = (__bf16)(s2[2] - d2[2]); a[11] = (__bf16)(s2[3] - d2[3]);
    a[12] = (__bf16)(s3[0] - d3[0]); a[13] = (__bf16)(s3[1] - d3[1]);
    a[14] = (__bf16)(s3[2] - d3[2]); a[15] = (__bf16)(s3[3] - d3[3]);

    // B 32x16 bf16 layout: lane = col f, K {8g..8g+7, 16+8g..16+8g+7} (matches A)
#pragma unroll
    for (int t = 0; t < 4; ++t) {
      const __bf16* xp = xtb + (size_t)(16 * t + n) * Nz + k0 + 8 * g;
      v16bf bt = cat8(*(const v8bf*)xp, *(const v8bf*)(xp + 16));
      acc[t] = __builtin_amdgcn_wmma_f32_16x16x32_bf16(
          false, a, false, bt, (short)0, acc[t], false, false);
    }
  }

  // Epilogue 1: y = x + acc/N in C layout (lane = col, rows i+8g); bf16 -> LDS
  const float invN = 1.0f / (float)Nz;
  __bf16* ytile = ldsY + wv * (16 * 64);
#pragma unroll
  for (int t = 0; t < 4; ++t) {
    const int col = 16 * t + n;
#pragma unroll
    for (int i = 0; i < 8; ++i) {
      const int r = i + 8 * g;
      float xv = node[((size_t)b * Nz + m0 + r) * Fz + col];
      ytile[r * 64 + col] = (__bf16)fmaf(acc[t][i], invN, xv);
    }
  }
  __syncthreads();

  // Epilogue 2: z = y @ W^T.  A from LDS (lane = row m), B from bf16 W rows.
  v8f zacc[4] = {v8f{}, v8f{}, v8f{}, v8f{}};
  const __bf16* yrow = ytile + n * 64;
#pragma unroll
  for (int s = 0; s < 2; ++s) {
    const __bf16* ap = yrow + 32 * s + 8 * g;
    v16bf a2 = cat8(*(const v8bf*)ap, *(const v8bf*)(ap + 16));
#pragma unroll
    for (int t = 0; t < 4; ++t) {
      const __bf16* wp = wb + (size_t)(16 * t + n) * Fz + 32 * s + 8 * g;
      v16bf b2 = cat8(*(const v8bf*)wp, *(const v8bf*)(wp + 16));
      zacc[t] = __builtin_amdgcn_wmma_f32_16x16x32_bf16(
          false, a2, false, b2, (short)0, zacc[t], false, false);
    }
  }

  // Store z + per-channel partial (sum, sumsq), reduced deterministically.
#pragma unroll
  for (int t = 0; t < 4; ++t) {
    const int col = 16 * t + n;
    float ls = 0.0f, lq = 0.0f;
#pragma unroll
    for (int i = 0; i < 8; ++i) {
      float v = zacc[t][i];
      z[((size_t)b * Nz + m0 + i + 8 * g) * Fz + col] = v;
      ls += v;
      lq += v * v;
    }
    ls += __shfl_xor(ls, 16, 32);   // combine row-halves (same channel)
    lq += __shfl_xor(lq, 16, 32);
    if (g == 0) { ldsSum[wv * 64 + col] = ls; ldsSq[wv * 64 + col] = lq; }
  }
  __syncthreads();
  if (wv == 0) {
    for (int c = ln; c < 64; c += 32) {
      float s = 0.0f, q = 0.0f;
#pragma unroll
      for (int w = 0; w < 8; ++w) { s += ldsSum[w * 64 + c]; q += ldsSq[w * 64 + c]; }
      psum[(size_t)blockIdx.x * 64 + c] = s;
      psq [(size_t)blockIdx.x * 64 + c] = q;
    }
  }
}

// k2: reduce 256 partials -> per-channel scale/shift (1 block, deterministic)
__global__ void k_stats(const float* __restrict__ psum, const float* __restrict__ psq,
                        const float* __restrict__ gamma, const float* __restrict__ beta,
                        float* __restrict__ ss) {
  const int ch = threadIdx.x;  // 64 threads
  float s = 0.0f, q = 0.0f;
  for (int w = 0; w < 256; ++w) { s += psum[w * 64 + ch]; q += psq[w * 64 + ch]; }
  const float inv = 1.0f / (float)(Bz * Nz);
  float mean = s * inv;
  float var  = fmaxf(q * inv - mean * mean, 0.0f);
  float scale = gamma[ch] * rsqrtf(var + BN_EPS);
  ss[ch]      = scale;
  ss[64 + ch] = beta[ch] - mean * scale;
}

// k3: out = relu(z*scale + shift), float4 vectorized (8MB in, 8MB out).
// out is never re-read -> non-temporal store.
__global__ __launch_bounds__(256) void k_bn_relu(
    const float* __restrict__ z, const float* __restrict__ ss,
    float* __restrict__ out) {
  size_t i = (size_t)blockIdx.x * 256 + threadIdx.x;  // float4 index
  const int o = (int)((i * 4) & 63);
  f4 zv = ((const f4*)z)[i];
  f4 sc = ((const f4*)ss)[o >> 2];
  f4 sh = ((const f4*)(ss + 64))[o >> 2];
  f4 r;
  r[0] = fmaxf(fmaf(zv[0], sc[0], sh[0]), 0.0f);
  r[1] = fmaxf(fmaf(zv[1], sc[1], sh[1]), 0.0f);
  r[2] = fmaxf(fmaf(zv[2], sc[2], sh[2]), 0.0f);
  r[3] = fmaxf(fmaf(zv[3], sc[3], sh[3]), 0.0f);
  __builtin_nontemporal_store(r, (f4*)out + i);
}

// ---------------------------------------------------------------------------
extern "C" void kernel_launch(void* const* d_in, const int* in_sizes, int n_in,
                              void* d_out, int out_size, void* d_ws, size_t ws_size,
                              hipStream_t stream) {
  (void)in_sizes; (void)n_in; (void)out_size; (void)ws_size;
  const float* node  = (const float*)d_in[0];   // (B,N,F)
  const float* edge  = (const float*)d_in[1];   // (B,2,N,N)
  const float* W     = (const float*)d_in[2];   // (F,F)
  const float* gamma = (const float*)d_in[3];
  const float* beta  = (const float*)d_in[4];
  float* out = (float*)d_out;

  // Workspace layout (~12.7 MB total)
  char* ws = (char*)d_ws;
  __bf16* xt  = (__bf16*)ws;                                   // 4 MB
  __bf16* wb  = (__bf16*)(ws + (size_t)4 * 1024 * 1024);       // 8 KB
  float* zbuf = (float*)(ws + (size_t)4 * 1024 * 1024 + 8192); // 8 MB
  float* psum = (float*)((char*)zbuf + (size_t)Bz * Nz * Fz * 4);
  float* psq  = psum + 256 * 64;
  float* ss   = psq + 256 * 64;

  k_transpose_x<<<Bz * (Nz / 32), 256, 0, stream>>>(node, xt);   // 1024 blocks
  k_convert_w<<<16, 256, 0, stream>>>(W, wb);
  k_fused_gemm<<<Bz * (Nz / 128), 256, 0, stream>>>(             // 256 blocks
      node, edge, xt, wb, zbuf, psum, psq);
  k_stats<<<1, 64, 0, stream>>>(psum, psq, gamma, beta, ss);
  k_bn_relu<<<(Bz * Nz * Fz) / 4 / 256, 256, 0, stream>>>(zbuf, ss, out); // 2048
}
